// Net_22565758173555
// MI455X (gfx1250) — matrix operand order repaired and checked by reading, not compiled, not run on hardware
//
#include <hip/hip_runtime.h>
#include <math.h>

// ---- CDNA5 WMMA types ----
typedef __attribute__((ext_vector_type(16))) _Float16 v16h;
typedef __attribute__((ext_vector_type(8)))  float    v8f;
typedef _Float16 v8h   __attribute__((ext_vector_type(8)));              // 16B-aligned
typedef _Float16 v8h_u __attribute__((ext_vector_type(8), aligned(2)));  // unaligned view

#define NB     16
#define HID    256
#define VOCAB  10000
#define TSTEPS 31
#define LFEAT  3136   // 56*56

__device__ __forceinline__ float sigmoidf_(float x) { return 1.f / (1.f + expf(-x)); }

// ---------------------------------------------------------------------------
// Fold BatchNorm into per-channel scale/bias:  y = conv*scale + bias, ReLU
// ---------------------------------------------------------------------------
__global__ void fold_bn_kernel(const float* __restrict__ b, const float* __restrict__ gamma,
                               const float* __restrict__ beta, const float* __restrict__ mean,
                               const float* __restrict__ var, float* __restrict__ scale,
                               float* __restrict__ bias, int C)
{
    int c = blockIdx.x * blockDim.x + threadIdx.x;
    if (c >= C) return;
    float s  = gamma[c] * rsqrtf(var[c] + 1e-5f);
    scale[c] = s;
    bias[c]  = (b[c] - mean[c]) * s + beta[c];
}

// Conv weights fp32 (N x K, OIHW flat) -> f16 (N x Kp), zero-padded K
__global__ void convw_f16_kernel(const float* __restrict__ W, _Float16* __restrict__ Wh,
                                 int N, int K, int Kp)
{
    int idx = blockIdx.x * blockDim.x + threadIdx.x;
    if (idx >= N * Kp) return;
    int n = idx / Kp, k = idx - n * Kp;
    Wh[idx] = (k < K) ? (_Float16)W[n * K + k] : (_Float16)0.f;
}

// Plain fp32 -> f16 copy
__global__ void dense_f16_kernel(const float* __restrict__ X, _Float16* __restrict__ Y, int n)
{
    int i = blockIdx.x * blockDim.x + threadIdx.x;
    if (i < n) Y[i] = (_Float16)X[i];
}

// images (16,3,224,224) fp32 -> padded f16 (16,3,226,226), zero border
__global__ void img_pad_kernel(const float* __restrict__ X, _Float16* __restrict__ Y)
{
    int idx = blockIdx.x * blockDim.x + threadIdx.x;
    const int P = 226, tot = NB * 3 * P * P;
    if (idx >= tot) return;
    int xx = idx % P, yy = (idx / P) % P, c = (idx / (P * P)) % 3, b = idx / (P * P * 3);
    float v = 0.f;
    if (xx >= 1 && xx <= 224 && yy >= 1 && yy <= 224)
        v = X[((b * 3 + c) * 224 + yy - 1) * 224 + xx - 1];
    Y[idx] = (_Float16)v;
}

__global__ void zero_h_kernel(_Float16* __restrict__ p, int n)
{
    int i = blockIdx.x * blockDim.x + threadIdx.x;
    if (i < n) p[i] = (_Float16)0.f;
}

__global__ void zero_f_kernel(float* __restrict__ p, int n)
{
    int i = blockIdx.x * blockDim.x + threadIdx.x;
    if (i < n) p[i] = 0.f;
}

// ---------------------------------------------------------------------------
// Implicit-GEMM conv + BN + ReLU with WMMA f16->f32.
// Input: padded f16 (Cin, H+2, W+2).  Weights: f16 (Cout x Kp), Kp%32==0.
// Block = 256 thr (8 waves) computes 64(M) x 32(N) tile; wave -> 16x16 subtile.
// Double-buffered LDS A panel: one barrier per K-step; next panel's global
// loads are issued right after the barrier so they overlap ds_load + wmma.
// FULLK=true drops the K-guard from the hot loop (all layers but the first).
// ---------------------------------------------------------------------------
template<int KS, bool FULLK>
__global__ __launch_bounds__(256) void conv_f16_wmma(
    const _Float16* __restrict__ Xp, const _Float16* __restrict__ Wh,
    const float* __restrict__ scale, const float* __restrict__ bias,
    _Float16* __restrict__ Yp, int Cin, int H, int W, int Cout, int K, int Kp)
{
    __shared__ __align__(16) _Float16 As[2][64][40];   // 2 panels, pitch 40 (bank spread)

    const int tid  = threadIdx.x;
    const int lane = tid & 31;
    const int wave = tid >> 5;
    const int row  = lane & 15;
    const int hi   = lane >> 4;
    const int Wp   = W + 2;
    const int off  = 1 - KS / 2;        // padded-coord offset (0 for 3x3, 1 for 1x1)

    // staging role: thread loads 8 consecutive-m halves for one k-column
    const int mgrp = tid & 7;
    const int kloc = tid >> 3;          // 0..31
    const int mbase = blockIdx.x * 64 + mgrp * 8;
    const int sx = mbase % W;
    const int sy = (mbase / W) % H;
    const int sbb = mbase / (W * H);

    // compute role
    const int m0 = blockIdx.x * 64 + (wave >> 1) * 16;
    const int n  = blockIdx.y * 32 + (wave & 1) * 16 + row;
    const _Float16* wrow = Wh + (size_t)n * Kp;

    auto stageLoad = [&](int k0) -> v8h_u {
        v8h_u a = {};
        int k = k0 + kloc;
        if (FULLK || k < K) {
            int ci, kh, kw;
            if (KS == 1) { ci = k; kh = 0; kw = 0; }
            else { kw = k % 3; kh = (k / 3) % 3; ci = k / 9; }
            const _Float16* src = Xp + ((size_t)(sbb * Cin + ci) * (H + 2) + (sy + kh + off)) * Wp
                                     + (sx + kw + off);
            a = *(const v8h_u*)src;
        }
        return a;
    };

    v8h_u a = stageLoad(0);
    v8f acc = {};
    for (int k0 = 0; k0 < Kp; k0 += 32) {
        const int buf = (k0 >> 5) & 1;
        // store current panel (its last readers synced one iteration ago)
#pragma unroll
        for (int j = 0; j < 8; ++j) As[buf][mgrp * 8 + j][kloc] = a[j];
        __syncthreads();
        if (k0 + 32 < Kp) a = stageLoad(k0 + 32);   // overlap with compute below

        const int r = ((wave >> 1) << 4) + row;
        v8h alo = *(const v8h*)&As[buf][r][hi * 8];
        v8h ahi = *(const v8h*)&As[buf][r][16 + hi * 8];
        v8h blo = *(const v8h*)(wrow + k0 + hi * 8);
        v8h bhi = *(const v8h*)(wrow + k0 + 16 + hi * 8);
        v16h af, bf;
#pragma unroll
        for (int i = 0; i < 8; ++i) {
            af[i] = alo[i]; af[i + 8] = ahi[i];
            bf[i] = blo[i]; bf[i + 8] = bhi[i];
        }
        acc = __builtin_amdgcn_wmma_f32_16x16x32_f16(false, af, false, bf,
                                                     (short)0, acc, false, false);
    }

    const float sc = scale[n];
    const float bs = bias[n];
#pragma unroll
    for (int j = 0; j < 8; ++j) {
        int mm = m0 + j + hi * 8;
        int xx = mm % W;
        int yy = (mm / W) % H;
        int bb = mm / (W * H);
        float v = acc[j] * sc + bs;
        v = v > 0.f ? v : 0.f;
        Yp[((size_t)(bb * Cout + n) * (H + 2) + yy + 1) * Wp + xx + 1] = (_Float16)v;
    }
}

// ---------------------------------------------------------------------------
// 2x2 max-pool f16, padded in (2H+2,2W+2) -> padded out (H+2,W+2), interior
// ---------------------------------------------------------------------------
__global__ void maxpool_f16_kernel(const _Float16* __restrict__ X, _Float16* __restrict__ Y,
                                   int C, int Ho, int Wo)
{
    int idx = blockIdx.x * blockDim.x + threadIdx.x;
    int total = NB * C * Ho * Wo;
    if (idx >= total) return;
    int x = idx % Wo, y = (idx / Wo) % Ho, c = (idx / (Wo * Ho)) % C, b = idx / (Wo * Ho * C);
    int Wi = 2 * Wo, bc = b * C + c;
    const _Float16* p = X + ((size_t)bc * (2 * Ho + 2) + 2 * y + 1) * (Wi + 2) + 2 * x + 1;
    float m = fmaxf(fmaxf((float)p[0], (float)p[1]),
                    fmaxf((float)p[Wi + 2], (float)p[Wi + 3]));
    Y[((size_t)bc * (Ho + 2) + y + 1) * (Wo + 2) + x + 1] = (_Float16)m;
}

// (B,256,58,58 padded f16) interior -> feats (B,3136,256) fp32
__global__ void to_feats_kernel(const _Float16* __restrict__ Xp, float* __restrict__ F)
{
    int idx = blockIdx.x * blockDim.x + threadIdx.x;
    int total = NB * LFEAT * HID;
    if (idx >= total) return;
    int h = idx % HID, l = (idx / HID) % LFEAT, b = idx / (HID * LFEAT);
    int y = l / 56, x = l - y * 56;
    F[idx] = (float)Xp[((size_t)(b * HID + h) * 58 + y + 1) * 58 + x + 1];
}

// ---------------------------------------------------------------------------
// Attention: s[b,l] = dot(feats[b,l,:], q[b,:]) ; one wave per (b,l)
// ---------------------------------------------------------------------------
__global__ __launch_bounds__(256) void attn_scores_kernel(
    const float* __restrict__ feats, const float* __restrict__ q, float* __restrict__ s)
{
    int wid  = (blockIdx.x * blockDim.x + threadIdx.x) >> 5;
    int lane = threadIdx.x & 31;
    if (wid >= NB * LFEAT) return;
    int b = wid / LFEAT, l = wid - b * LFEAT;
    const float* f  = feats + ((size_t)b * LFEAT + l) * HID;
    const float* qq = q + b * HID;
    float acc = 0.f;
    for (int k = lane; k < HID; k += 32) acc += f[k] * qq[k];
#pragma unroll
    for (int off = 16; off; off >>= 1) acc += __shfl_xor(acc, off, 32);
    if (lane == 0) s[wid] = acc;
}

__global__ __launch_bounds__(256) void softmax_kernel(float* __restrict__ s, int L)
{
    __shared__ float red[256];
    int b = blockIdx.x;
    float m = -INFINITY;
    for (int l = threadIdx.x; l < L; l += 256) m = fmaxf(m, s[b * L + l]);
    red[threadIdx.x] = m; __syncthreads();
    for (int off = 128; off; off >>= 1) {
        if (threadIdx.x < off) red[threadIdx.x] = fmaxf(red[threadIdx.x], red[threadIdx.x + off]);
        __syncthreads();
    }
    m = red[0]; __syncthreads();
    float sum = 0.f;
    for (int l = threadIdx.x; l < L; l += 256) {
        float e = expf(s[b * L + l] - m);
        s[b * L + l] = e;
        sum += e;
    }
    red[threadIdx.x] = sum; __syncthreads();
    for (int off = 128; off; off >>= 1) {
        if (threadIdx.x < off) red[threadIdx.x] += red[threadIdx.x + off];
        __syncthreads();
    }
    float inv = 1.f / red[0];
    for (int l = threadIdx.x; l < L; l += 256) s[b * L + l] *= inv;
}

// ctx[b,k] = sum_l w[b,l]*feats[b,l,k]; coalesced across k, prefetch ahead on l
__global__ __launch_bounds__(256) void attn_ctx_kernel(
    const float* __restrict__ feats, const float* __restrict__ w, float* __restrict__ ctx)
{
    int b = blockIdx.x, k = threadIdx.x;
    const float* f  = feats + (size_t)b * LFEAT * HID + k;
    const float* ww = w + b * LFEAT;
    float acc = 0.f;
    for (int l = 0; l < LFEAT; ++l) {
        if ((l & 7) == 0 && l + 8 < LFEAT)
            __builtin_prefetch(f + (size_t)(l + 8) * HID, 0, 0);   // global_prefetch_b8
        acc += ww[l] * f[(size_t)l * HID];
    }
    ctx[b * HID + k] = acc;
}

// x_t = concat(embed_f16[captions[:,t]], (f16)ctx)  -> (16,512) f16
__global__ void build_x_kernel(const _Float16* __restrict__ embh, const int* __restrict__ captions,
                               const float* __restrict__ ctx, _Float16* __restrict__ x, int t)
{
    int idx = blockIdx.x * blockDim.x + threadIdx.x;
    if (idx >= NB * 2 * HID) return;
    int b = idx >> 9, k = idx & 511;
    _Float16 v;
    if (k < HID) {
        int tok = captions[b * 32 + t];
        v = embh[(size_t)tok * HID + k];
    } else {
        v = (_Float16)ctx[b * HID + (k - HID)];
    }
    x[idx] = v;
}

// ---------------------------------------------------------------------------
// Dense GEMM, M fixed = 16:  Out[m, ooff+n] (+)= A(16xK) * B(NxK)^T + bias[n]
// f16 operands, branch-free aligned 16B loads, one wave per 16-wide N tile.
// ---------------------------------------------------------------------------
__global__ __launch_bounds__(128) void gemm16_f16_wmma(
    const _Float16* __restrict__ A, int lda,
    const _Float16* __restrict__ Bm, int ldb,
    const float* __restrict__ bias,
    float* __restrict__ Out, int ldo, int ooff,
    int N, int K, int accumulate)
{
    const int lane  = threadIdx.x & 31;
    const int wave  = threadIdx.x >> 5;
    const int ntile = blockIdx.x * 4 + wave;
    if (ntile * 16 >= N) return;
    const int row = lane & 15, hi = lane >> 4;
    const int n = ntile * 16 + row;
    const _Float16* arow = A + row * lda;
    const _Float16* brow = Bm + (size_t)n * ldb;

    v8f acc = {};
    for (int k0 = 0; k0 < K; k0 += 32) {
        v8h alo = *(const v8h*)(arow + k0 + hi * 8);
        v8h ahi = *(const v8h*)(arow + k0 + 16 + hi * 8);
        v8h blo = *(const v8h*)(brow + k0 + hi * 8);
        v8h bhi = *(const v8h*)(brow + k0 + 16 + hi * 8);
        v16h af, bf;
#pragma unroll
        for (int i = 0; i < 8; ++i) {
            af[i] = alo[i]; af[i + 8] = ahi[i];
            bf[i] = blo[i]; bf[i + 8] = bhi[i];
        }
        acc = __builtin_amdgcn_wmma_f32_16x16x32_f16(false, af, false, bf,
                                                     (short)0, acc, false, false);
    }
#pragma unroll
    for (int j = 0; j < 8; ++j) {
        int mm = j + hi * 8;
        float v = acc[j] + bias[n];
        if (accumulate) Out[mm * ldo + ooff + n] += v;
        else            Out[mm * ldo + ooff + n]  = v;
    }
}

__global__ void h_f16_kernel(const float* __restrict__ h, _Float16* __restrict__ hh)
{
    int i = blockIdx.x * blockDim.x + threadIdx.x;
    if (i < NB * HID) hh[i] = (_Float16)h[i];
}

__global__ void lstm_cell_kernel(const float* __restrict__ gates,
                                 float* __restrict__ h, float* __restrict__ c)
{
    int idx = blockIdx.x * blockDim.x + threadIdx.x;
    if (idx >= NB * HID) return;
    int b = idx >> 8, j = idx & 255;
    const float* g = gates + b * 4 * HID;
    float ig = sigmoidf_(g[j]);
    float fg = sigmoidf_(g[HID + j]);
    float gg = tanhf(g[2 * HID + j]);
    float og = sigmoidf_(g[3 * HID + j]);
    float cn = fg * c[idx] + ig * gg;
    c[idx] = cn;
    h[idx] = og * tanhf(cn);
}

__global__ void copy_hc_kernel(const float* __restrict__ h, const float* __restrict__ c,
                               float* __restrict__ out)
{
    int i = blockIdx.x * blockDim.x + threadIdx.x;
    if (i >= 2 * NB * HID) return;
    out[i] = (i < NB * HID) ? h[i] : c[i - NB * HID];
}

// ---------------------------------------------------------------------------
// Host orchestration
// ---------------------------------------------------------------------------
extern "C" void kernel_launch(void* const* d_in, const int* in_sizes, int n_in,
                              void* d_out, int out_size, void* d_ws, size_t ws_size,
                              hipStream_t stream)
{
    const float* images   = (const float*)d_in[0];
    const int*   captions = (const int*)d_in[1];
    const float *cw[7], *cb[7], *cg[7], *cbe[7], *cme[7], *cva[7];
    for (int i = 0; i < 7; ++i) {
        int base = 2 + i * 6;
        cw[i]  = (const float*)d_in[base + 0];
        cb[i]  = (const float*)d_in[base + 1];
        cg[i]  = (const float*)d_in[base + 2];
        cbe[i] = (const float*)d_in[base + 3];
        cme[i] = (const float*)d_in[base + 4];
        cva[i] = (const float*)d_in[base + 5];
    }
    const float* embed = (const float*)d_in[44];
    const float* Wq    = (const float*)d_in[45];
    const float* bq    = (const float*)d_in[46];
    const float* W_ih  = (const float*)d_in[47];
    const float* b_ih  = (const float*)d_in[48];
    const float* W_hh  = (const float*)d_in[49];
    const float* b_hh  = (const float*)d_in[50];
    const float* Wfc   = (const float*)d_in[51];
    const float* bfc   = (const float*)d_in[52];
    float* out = (float*)d_out;
    (void)in_sizes; (void)n_in; (void)out_size; (void)ws_size;

    // ---- workspace carve-out ----
    char* ws = (char*)d_ws;
    size_t off = 0;
    auto alloc = [&](size_t bytes) { size_t o = off; off += (bytes + 255) & ~(size_t)255; return o; };
    const size_t padHalves = (size_t)NB * 64 * 226 * 226;   // largest padded activation
    _Float16* bufA = (_Float16*)(ws + alloc(padHalves * 2));
    _Float16* bufB = (_Float16*)(ws + alloc(padHalves * 2));
    float*    feats = (float*)(ws + alloc((size_t)NB * LFEAT * HID * 4));
    const int cins[7]  = {3, 64, 64, 128, 128, 256, 256};
    const int couts[7] = {64, 64, 128, 128, 256, 256, 256};
    const int kss[7]   = {3, 3, 3, 3, 3, 3, 1};
    int Kc[7], Kpc[7];
    _Float16* wh[7];
    for (int l = 0; l < 7; ++l) {
        Kc[l]  = cins[l] * kss[l] * kss[l];
        Kpc[l] = (Kc[l] + 31) & ~31;
        wh[l]  = (_Float16*)(ws + alloc((size_t)couts[l] * Kpc[l] * 2));
    }
    _Float16* wqh  = (_Float16*)(ws + alloc((size_t)HID * HID * 2));
    _Float16* wihh = (_Float16*)(ws + alloc((size_t)4 * HID * 2 * HID * 2));
    _Float16* whhh = (_Float16*)(ws + alloc((size_t)4 * HID * HID * 2));
    _Float16* wfch = (_Float16*)(ws + alloc((size_t)VOCAB * HID * 2));
    _Float16* embh = (_Float16*)(ws + alloc((size_t)VOCAB * HID * 2));
    float* sb    = (float*)(ws + alloc(7 * 512 * 4));
    float* qbuf  = (float*)(ws + alloc(NB * HID * 4));
    float* sbuf  = (float*)(ws + alloc((size_t)NB * LFEAT * 4));
    float* ctxb  = (float*)(ws + alloc(NB * HID * 4));
    float* gates = (float*)(ws + alloc(NB * 4 * HID * 4));
    float* hbuf  = (float*)(ws + alloc(NB * HID * 4));
    float* cbuf  = (float*)(ws + alloc(NB * HID * 4));
    _Float16* xh = (_Float16*)(ws + alloc(NB * 2 * HID * 2));
    _Float16* hh = (_Float16*)(ws + alloc(NB * HID * 2));

    // ---- pre-pass: BN fold + weight conversions ----
    for (int l = 0; l < 7; ++l) {
        fold_bn_kernel<<<(couts[l] + 255) / 256, 256, 0, stream>>>(
            cb[l], cg[l], cbe[l], cme[l], cva[l], sb + l * 512, sb + l * 512 + 256, couts[l]);
        int tot = couts[l] * Kpc[l];
        convw_f16_kernel<<<(tot + 255) / 256, 256, 0, stream>>>(cw[l], wh[l], couts[l], Kc[l], Kpc[l]);
    }
    dense_f16_kernel<<<(HID * HID + 255) / 256, 256, 0, stream>>>(Wq, wqh, HID * HID);
    dense_f16_kernel<<<(4 * HID * 2 * HID + 255) / 256, 256, 0, stream>>>(W_ih, wihh, 4 * HID * 2 * HID);
    dense_f16_kernel<<<(4 * HID * HID + 255) / 256, 256, 0, stream>>>(W_hh, whhh, 4 * HID * HID);
    dense_f16_kernel<<<(VOCAB * HID + 255) / 256, 256, 0, stream>>>(Wfc, wfch, VOCAB * HID);
    dense_f16_kernel<<<(VOCAB * HID + 255) / 256, 256, 0, stream>>>(embed, embh, VOCAB * HID);

    // ---- encoder ----
    auto zeroH = [&](_Float16* p, size_t n) {
        zero_h_kernel<<<(int)((n + 255) / 256), 256, 0, stream>>>(p, (int)n);
    };
    auto conv = [&](const _Float16* X, int l, _Float16* Y, int H) {
        int M = NB * H * H;
        dim3 grid(M / 64, couts[l] / 32);
        const float* sc = sb + l * 512;
        const float* bs = sb + l * 512 + 256;
        if (kss[l] == 3) {
            if (Kc[l] == Kpc[l])
                conv_f16_wmma<3, true><<<grid, 256, 0, stream>>>(X, wh[l], sc, bs, Y,
                    cins[l], H, H, couts[l], Kc[l], Kpc[l]);
            else
                conv_f16_wmma<3, false><<<grid, 256, 0, stream>>>(X, wh[l], sc, bs, Y,
                    cins[l], H, H, couts[l], Kc[l], Kpc[l]);
        } else {
            conv_f16_wmma<1, true><<<grid, 256, 0, stream>>>(X, wh[l], sc, bs, Y,
                cins[l], H, H, couts[l], Kc[l], Kpc[l]);
        }
    };
    auto pool = [&](const _Float16* X, _Float16* Y, int C, int Ho) {
        int total = NB * C * Ho * Ho;
        maxpool_f16_kernel<<<(total + 255) / 256, 256, 0, stream>>>(X, Y, C, Ho, Ho);
    };

    {
        int tot = NB * 3 * 226 * 226;
        img_pad_kernel<<<(tot + 255) / 256, 256, 0, stream>>>(images, bufB);
    }
    zeroH(bufA, (size_t)NB * 64 * 226 * 226);           // conv0 out feeds 3x3 conv1
    conv(bufB, 0, bufA, 224);
    conv(bufA, 1, bufB, 224);                           // -> pool (interior reads)
    zeroH(bufA, (size_t)NB * 64 * 114 * 114);           // pool1 out feeds 3x3 conv2
    pool(bufB, bufA, 64, 112);
    zeroH(bufB, (size_t)NB * 128 * 114 * 114);          // conv2 out feeds 3x3 conv3
    conv(bufA, 2, bufB, 112);
    conv(bufB, 3, bufA, 112);                           // -> pool
    zeroH(bufB, (size_t)NB * 128 * 58 * 58);            // pool2 out feeds 3x3 conv4
    pool(bufA, bufB, 128, 56);
    zeroH(bufA, (size_t)NB * 256 * 58 * 58);            // conv4 out feeds 3x3 conv5
    conv(bufB, 4, bufA, 56);
    conv(bufA, 5, bufB, 56);                            // -> 1x1 conv (interior reads)
    conv(bufB, 6, bufA, 56);                            // -> transpose (interior reads)
    {
        int tot = NB * LFEAT * HID;
        to_feats_kernel<<<(tot + 255) / 256, 256, 0, stream>>>(bufA, feats);
    }

    // ---- decoder ----
    zero_f_kernel<<<(NB * HID + 255) / 256, 256, 0, stream>>>(hbuf, NB * HID);
    zero_f_kernel<<<(NB * HID + 255) / 256, 256, 0, stream>>>(cbuf, NB * HID);

    const int scoreWaves = NB * LFEAT;
    for (int t = 0; t < TSTEPS; ++t) {
        h_f16_kernel<<<(NB * HID + 255) / 256, 256, 0, stream>>>(hbuf, hh);
        // q = h @ Wq^T + bq
        gemm16_f16_wmma<<<(HID / 16 + 3) / 4, 128, 0, stream>>>(
            hh, HID, wqh, HID, bq, qbuf, HID, 0, HID, HID, 0);
        // attention
        attn_scores_kernel<<<(scoreWaves * 32 + 255) / 256, 256, 0, stream>>>(feats, qbuf, sbuf);
        softmax_kernel<<<NB, 256, 0, stream>>>(sbuf, LFEAT);
        attn_ctx_kernel<<<NB, 256, 0, stream>>>(feats, sbuf, ctxb);
        // x = [emb_t, ctx] (f16)
        build_x_kernel<<<(NB * 2 * HID + 255) / 256, 256, 0, stream>>>(embh, captions, ctxb, xh, t);
        // gates = x @ W_ih^T + b_ih + h @ W_hh^T + b_hh
        gemm16_f16_wmma<<<(4 * HID / 16 + 3) / 4, 128, 0, stream>>>(
            xh, 2 * HID, wihh, 2 * HID, b_ih, gates, 4 * HID, 0, 4 * HID, 2 * HID, 0);
        gemm16_f16_wmma<<<(4 * HID / 16 + 3) / 4, 128, 0, stream>>>(
            hh, HID, whhh, HID, b_hh, gates, 4 * HID, 0, 4 * HID, HID, 1);
        // cell update
        lstm_cell_kernel<<<(NB * HID + 255) / 256, 256, 0, stream>>>(gates, hbuf, cbuf);
        // logits_t = h_new @ Wfc^T + bfc
        h_f16_kernel<<<(NB * HID + 255) / 256, 256, 0, stream>>>(hbuf, hh);
        gemm16_f16_wmma<<<(VOCAB / 16 + 3) / 4, 128, 0, stream>>>(
            hh, HID, wfch, HID, bfc, out, TSTEPS * VOCAB, t * VOCAB, VOCAB, HID, 0);
    }

    copy_hc_kernel<<<(2 * NB * HID + 255) / 256, 256, 0, stream>>>(
        hbuf, cbuf, out + (size_t)NB * TSTEPS * VOCAB);
}
